// IrisSpecializedLossV3_32555852103877
// MI455X (gfx1250) — 8-bit matrix operand order repaired and checked by reading, compiled, not run on hardware
//
#include <hip/hip_runtime.h>
#include <hip/hip_bf16.h>

typedef int   v8i __attribute__((ext_vector_type(8)));
typedef float v2f __attribute__((ext_vector_type(2)));
typedef int   v2i __attribute__((ext_vector_type(2)));

#define NB 4096
#define NC 10
#define HW 900
#define PADN 960

__device__ __forceinline__ unsigned lds_off(const void* p) {
    return (unsigned)(size_t)p;
}

__global__ __launch_bounds__(256) void iris_loss_main(
    const float* __restrict__ pred, const int* __restrict__ tgt,
    const int* __restrict__ inp, float* __restrict__ ws)
{
    __shared__ alignas(16) unsigned char s_in[PADN];
    __shared__ alignas(16) unsigned char s_tg[PADN];
    __shared__ alignas(16) unsigned char s_pd[PADN];
    __shared__ int   s_cnt_t[256];
    __shared__ int   s_cnt_p[256];
    __shared__ float s_red[8][4];

    const int b    = blockIdx.x;
    const int t    = threadIdx.x;
    const int wave = t >> 5;
    const int lane = t & 31;

    const unsigned long long inB = (unsigned long long)(size_t)(inp + b * HW);
    const unsigned long long tgB = (unsigned long long)(size_t)(tgt + b * HW);
    for (int p = t; p < HW; p += 256) {
        unsigned li = lds_off(&s_in[p]);
        unsigned lt = lds_off(&s_tg[p]);
        unsigned long long ga = inB + (unsigned long long)p * 4ull;
        unsigned long long gt = tgB + (unsigned long long)p * 4ull;
        asm volatile("global_load_async_to_lds_b8 %0, %1, off" :: "v"(li), "v"(ga) : "memory");
        asm volatile("global_load_async_to_lds_b8 %0, %1, off" :: "v"(lt), "v"(gt) : "memory");
    }
    if (t < PADN - HW) { s_in[HW + t] = 15; s_tg[HW + t] = 15; s_pd[HW + t] = 15; }

    float f_focal = 0.f; int f_eq = 0, f_cp = 0, f_nc = 0;
    const float* pb = pred + (size_t)b * (NC * HW);
    for (int q = t; q < HW / 2; q += 256) {
        const int p = q * 2;
        float xs0[NC], xs1[NC];
        float mx0 = -3.4e38f, mx1 = -3.4e38f, sx0 = 0.f, sx1 = 0.f;
        int am0 = 0, am1 = 0;
        #pragma unroll
        for (int c = 0; c < NC; ++c) {
            v2f x = __builtin_nontemporal_load((const v2f*)(pb + c * HW + p));
            xs0[c] = x.x; xs1[c] = x.y; sx0 += x.x; sx1 += x.y;
            if (x.x > mx0) { mx0 = x.x; am0 = c; }
            if (x.y > mx1) { mx1 = x.y; am1 = c; }
        }
        float se0 = 0.f, se1 = 0.f;
        #pragma unroll
        for (int c = 0; c < NC; ++c) { se0 += __expf(xs0[c] - mx0); se1 += __expf(xs1[c] - mx1); }
        float lse0 = mx0 + __logf(se0);
        float lse1 = mx1 + __logf(se1);
        v2i tg2 = *(const v2i*)(tgt + b * HW + p);
        v2i in2 = *(const v2i*)(inp + b * HW + p);
        float xt0 = xs0[0], xt1 = xs1[0];
        #pragma unroll
        for (int c = 1; c < NC; ++c) {
            if (tg2.x == c) xt0 = xs0[c];
            if (tg2.y == c) xt1 = xs1[c];
        }
        f_focal += (lse0 - 0.9f * xt0 - 0.01f * sx0) + (lse1 - 0.9f * xt1 - 0.01f * sx1);
        f_eq += (am0 == tg2.x) + (am1 == tg2.y);
        f_cp += (am0 == in2.x) + (am1 == in2.y);
        f_nc += (tg2.x != in2.x) + (tg2.y != in2.y);
        *(unsigned short*)&s_pd[p] = (unsigned short)(am0 | (am1 << 8));
    }
    #pragma unroll
    for (int o = 16; o > 0; o >>= 1) {
        f_focal += __shfl_xor(f_focal, o, 32);
        f_eq    += __shfl_xor(f_eq, o, 32);
        f_cp    += __shfl_xor(f_cp, o, 32);
        f_nc    += __shfl_xor(f_nc, o, 32);
    }
    if (lane == 0) {
        s_red[wave][0] = f_focal; s_red[wave][1] = (float)f_eq;
        s_red[wave][2] = (float)f_cp; s_red[wave][3] = (float)f_nc;
    }

    asm volatile("s_wait_asynccnt 0" ::: "memory");
    __syncthreads();

    if (wave < 2) {
        const unsigned mn = (unsigned)lane & 15u;
        const unsigned hs = (unsigned)lane >> 4;
        const unsigned repl = mn * 0x01010101u;
        const unsigned char* Bsrc = wave ? s_pd : s_tg;
        v8i acc = {};
        for (int ch = 0; ch < 15; ++ch) {
            const unsigned char* ab = s_in + ch * 64;
            const unsigned char* bb = Bsrc + ch * 64;
            v8i A, Bm;
            #pragma unroll
            for (int v = 0; v < 8; ++v) {
                unsigned kA = (((unsigned)v >> 2) << 5) | (hs << 4) | (((unsigned)v & 3u) << 2);
                unsigned x = *(const unsigned*)(ab + kA) ^ repl;
                unsigned u = x | (x >> 1); u |= (u >> 2);
                A[v] = (int)(~u & 0x01010101u);
                unsigned kB = (((unsigned)v >> 2) << 5) | (hs << 4) | (((unsigned)v & 3u) << 2);
                unsigned y = *(const unsigned*)(bb + kB) ^ repl;
                unsigned w2 = y | (y >> 1); w2 |= (w2 >> 2);
                Bm[v] = (int)(~w2 & 0x01010101u);
            }
            acc = __builtin_amdgcn_wmma_i32_16x16x64_iu8(false, A, false, Bm, acc, false, false);
        }
        int* cnt = wave ? s_cnt_p : s_cnt_t;
        #pragma unroll
        for (int r = 0; r < 8; ++r) cnt[(r + 8 * (int)hs) * 16 + (int)mn] = acc[r];
    }
    __syncthreads();

    if (t == 0) {
        float focal = 0.f, eq = 0.f, cp = 0.f, nc = 0.f;
        for (int w = 0; w < 8; ++w) {
            focal += s_red[w][0]; eq += s_red[w][1]; cp += s_red[w][2]; nc += s_red[w][3];
        }
        int n_t = 0, n_p = 0;
        for (int n = 0; n < NC; ++n) {
            int st = 0, sp = 0;
            for (int m = 0; m < NC; ++m) { st += s_cnt_t[m * 16 + n]; sp += s_cnt_p[m * 16 + n]; }
            n_t += (st > 0); n_p += (sp > 0);
        }
        int sB = 0, nB = 0;
        for (int m = 0; m < NC; ++m) {
            int rsum = 0, tm = 0, pm = 0, tv = -1, pv = -1;
            for (int n = 0; n < NC; ++n) {
                int ct = s_cnt_t[m * 16 + n], cq = s_cnt_p[m * 16 + n];
                rsum += ct;
                if (ct > tv) { tv = ct; tm = n; }
                if (cq > pv) { pv = cq; pm = n; }
            }
            if (rsum > 0) { nB++; sB += (tm == pm); }
        }
        int nb = (nB > 0) ? nB : 1;
        float* w = ws + b * 8;
        w[0] = focal;
        w[1] = eq;
        w[2] = (eq == 900.f) ? 1.f : 0.f;
        w[3] = (cp == 900.f) ? 1.f : 0.f;
        w[4] = nc * (1.0f / 900.f);
        w[5] = __expf(-0.5f * fabsf((float)(n_p - n_t)));
        w[6] = (float)sB;
        w[7] = (float)nb;
    }
}

__global__ __launch_bounds__(256) void iris_loss_finalize(
    const float* __restrict__ ws, float* __restrict__ out)
{
    __shared__ float r0[256], r1[256], r2[256], r3[256], r4[256], r5[256];
    __shared__ float rA[256], rC[256];
    const int t = threadIdx.x;

    float s_focal = 0.f, s_eq = 0.f, s_ex = 0.f, s_cp = 0.f, s_pat = 0.f, s_har = 0.f;
    for (int b = t; b < NB; b += 256) {
        const float* w = ws + b * 8;
        s_focal += w[0];
        float eqc = w[1];
        s_eq += eqc;
        s_ex += w[2];
        s_cp += w[3];
        s_pat += (eqc * (1.0f / 900.0f)) * (1.0f + 0.5f * w[4]);
        s_har += w[5];
    }
    float A = 1.0f, C = 0.0f;
    const int b0 = t * 16;
    for (int i = 0; i < 16; ++i) {
        float s = ws[(b0 + i) * 8 + 6];
        float n = ws[(b0 + i) * 8 + 7];
        A = A / n;
        C = (C + s) / n;
    }
    r0[t] = s_focal; r1[t] = s_eq; r2[t] = s_ex; r3[t] = s_cp; r4[t] = s_pat; r5[t] = s_har;
    rA[t] = A; rC[t] = C;
    __syncthreads();

    for (int cnt = 256; cnt > 1; cnt >>= 1) {
        const int half = cnt >> 1;
        const bool act = (t < half);
        float v0 = 0, v1 = 0, v2 = 0, v3 = 0, v4 = 0, v5 = 0;
        float Al = 1, Cl = 0, Ar = 1, Cr = 0;
        if (act) {
            v0 = r0[2 * t] + r0[2 * t + 1]; v1 = r1[2 * t] + r1[2 * t + 1];
            v2 = r2[2 * t] + r2[2 * t + 1]; v3 = r3[2 * t] + r3[2 * t + 1];
            v4 = r4[2 * t] + r4[2 * t + 1]; v5 = r5[2 * t] + r5[2 * t + 1];
            Al = rA[2 * t]; Cl = rC[2 * t]; Ar = rA[2 * t + 1]; Cr = rC[2 * t + 1];
        }
        __syncthreads();
        if (act) {
            r0[t] = v0; r1[t] = v1; r2[t] = v2; r3[t] = v3; r4[t] = v4; r5[t] = v5;
            rA[t] = Al * Ar; rC[t] = Ar * Cl + Cr;
        }
        __syncthreads();
    }

    if (t == 0) {
        const float invB = 1.0f / (float)NB;
        float focal      = r0[0] / ((float)NB * 900.0f);
        float iou_mean   = r1[0] * invB * (1.0f / 900.0f);
        float exact_mean = r2[0] * invB;
        float combined   = 0.15f * exact_mean + 0.85f * iou_mean;
        float exact_bonus = fmaxf(-combined * 5.0f, -4.0f);
        float transform  = (r3[0] * invB) * 0.5f;
        float pattern    = -(r4[0] * invB) * 0.1f * 0.2f;
        float harmony    = -(r5[0] * invB) * 0.05f * 0.15f;
        float transition = rC[0] * invB;
        float trans_bonus = -transition * 0.08f * 0.1f;
        out[0] = focal + transform + exact_bonus + pattern + harmony + trans_bonus;
    }
}

extern "C" void kernel_launch(void* const* d_in, const int* in_sizes, int n_in,
                              void* d_out, int out_size, void* d_ws, size_t ws_size,
                              hipStream_t stream) {
    const float* pred = (const float*)d_in[0];
    const int*   tgts = (const int*)d_in[1];
    const int*   inps = (const int*)d_in[2];
    float* out = (float*)d_out;
    float* ws  = (float*)d_ws;
    iris_loss_main<<<NB, 256, 0, stream>>>(pred, tgts, inps, ws);
    iris_loss_finalize<<<1, 256, 0, stream>>>(ws, out);
}
